// OrthoLinear_43112881717934
// MI455X (gfx1250) — compile-verified, matrix operand order checked by reading, not run on hardware
//
#include <hip/hip_runtime.h>

typedef _Float16 half8 __attribute__((ext_vector_type(8)));
typedef _Float16 v16h  __attribute__((ext_vector_type(16)));
typedef float    v8f   __attribute__((ext_vector_type(8)));
typedef float    f4v   __attribute__((ext_vector_type(4)));

#define IN_F   4096
#define OUT_F  4096
#define BM     256   // block tile M
#define BN     128   // block tile N
#define TK     32    // k-step per WMMA
#define NT     (IN_F / TK)   // 128 k-tiles
#define LDA    40    // halves per LDS row (80B): conflict-free b128 fragment loads

#define LDSA_HALVES (BM * LDA)                  // 10240 halves = 20480 B
#define LDSB_HALVES (BN * LDA)                  //  5120 halves = 10240 B
#define BUF_HALVES  (LDSA_HALVES + LDSB_HALVES) // 15360 halves = 30720 B
#define BUF_BYTES   (BUF_HALVES * 2)

// ---------------- x fp32 -> fp16 (one pass, halves A-stream traffic) ----------
__global__ __launch_bounds__(256) void cvt_x_kernel(const float* __restrict__ x,
                                                    _Float16* __restrict__ x16,
                                                    int n8) {
  int i = blockIdx.x * blockDim.x + threadIdx.x;
  if (i >= n8) return;
  const f4v* src = (const f4v*)x;
  f4v a = src[2 * i], b = src[2 * i + 1];
  half8 h = { (_Float16)a.x, (_Float16)a.y, (_Float16)a.z, (_Float16)a.w,
              (_Float16)b.x, (_Float16)b.y, (_Float16)b.z, (_Float16)b.w };
  ((half8*)x16)[i] = h;
}

// ---------------- W_comb = base (vectorized 16B copy) -------------------------
__global__ __launch_bounds__(256) void copy_w_kernel(const f4v* __restrict__ src,
                                                     f4v* __restrict__ dst, int n) {
  int i = blockIdx.x * blockDim.x + threadIdx.x;
  if (i < n) dst[i] = src[i];
}

// ---------------- scatter-add sparse ortho into W_comb (fp16 via 32b CAS) -----
__global__ __launch_bounds__(256) void scatter_kernel(const _Float16* __restrict__ vals,
                                                      const int* __restrict__ rows,
                                                      const int* __restrict__ cols,
                                                      unsigned int* __restrict__ w,
                                                      int nnz) {
  int i = blockIdx.x * blockDim.x + threadIdx.x;
  if (i >= nnz) return;
  int idx = rows[i] * IN_F + cols[i];
  unsigned int wi = (unsigned)idx >> 1;
  int hi = idx & 1;
  float v = (float)vals[i];
  unsigned int old = w[wi];
  for (;;) {
    unsigned short hb = hi ? (unsigned short)(old >> 16) : (unsigned short)(old & 0xffffu);
    _Float16 cur;
    __builtin_memcpy(&cur, &hb, 2);
    _Float16 sum = (_Float16)((float)cur + v);
    unsigned short sb;
    __builtin_memcpy(&sb, &sum, 2);
    unsigned int neu = hi ? ((old & 0x0000ffffu) | ((unsigned)sb << 16))
                          : ((old & 0xffff0000u) | (unsigned)sb);
    unsigned int prev = atomicCAS(&w[wi], old, neu);
    if (prev == old) break;
    old = prev;
  }
}

// ---- async global->LDS copy of one 16B chunk per lane (CDNA5, ASYNCcnt) ------
__device__ __forceinline__ void async_b128(unsigned lds_byte_addr, unsigned gbl_byte_off,
                                           const void* sbase) {
  asm volatile("global_load_async_to_lds_b128 %0, %1, %2"
               :: "v"(lds_byte_addr), "v"(gbl_byte_off), "s"(sbase)
               : "memory");
}
__device__ __forceinline__ void wait_async(void) {   // all done
  asm volatile("s_wait_asynccnt 0x0" ::: "memory");
}
__device__ __forceinline__ void wait_async6(void) {  // <=6 outstanding (1 tile in flight)
  asm volatile("s_wait_asynccnt 0x6" ::: "memory");
}

#define LOADFRAG(p) __builtin_shufflevector(*(const half8*)(p), *(const half8*)((p) + 16), \
                                            0,1,2,3,4,5,6,7,8,9,10,11,12,13,14,15)

// ---------------- main f16 WMMA GEMM: out[M,N] = X16 @ Wcomb^T ----------------
// Block: 256 thr = 8 waves; block tile 256x128; wave tile 32x128 (2 M-frags).
// Triple-buffered LDS: tile k computes while k+1 and k+2 stream in on ASYNCcnt,
// giving ~2 tiles (32 WMMAs) of DMA latency slack. B fragments double-buffered
// in VGPRs so ds_load latency is overlapped with the WMMA pair of the previous
// fragment.
__global__ __launch_bounds__(256) void gemm_wmma_kernel(const _Float16* __restrict__ A,
                                                        const _Float16* __restrict__ B,
                                                        float* __restrict__ out) {
  __shared__ __align__(16) _Float16 smem[3 * BUF_HALVES];   // 92160 B

  const int tid   = threadIdx.x;
  const int lane  = tid & 31;             // wave32
  const int wave  = tid >> 5;             // 8 waves/block
  const int tileN = blockIdx.x * BN;
  const int tileM = blockIdx.y * BM;

  v8f acc0[8], acc1[8];
#pragma unroll
  for (int f = 0; f < 8; ++f) {
    acc0[f] = (v8f){0.f,0.f,0.f,0.f,0.f,0.f,0.f,0.f};
    acc1[f] = (v8f){0.f,0.f,0.f,0.f,0.f,0.f,0.f,0.f};
  }

  // Staging map: rows of 32 halves = 4 chunks of 16B.
  // A: 256 rows -> 4 chunks/thread; B: 128 rows -> 2 chunks/thread.
  const int r = tid >> 2;
  const int c = (tid & 3) * 8;            // in halves

  const _Float16* sbaseA = A + (size_t)tileM * IN_F;   // block-uniform SGPR base
  const _Float16* sbaseB = B + (size_t)tileN * IN_F;

  // per-lane global byte offsets for the 6 chunks (advance +64B per k-tile)
  unsigned offA0 = (unsigned)(((r      ) * IN_F + c) * 2);
  unsigned offA1 = (unsigned)(((r +  64) * IN_F + c) * 2);
  unsigned offA2 = (unsigned)(((r + 128) * IN_F + c) * 2);
  unsigned offA3 = (unsigned)(((r + 192) * IN_F + c) * 2);
  unsigned offB0 = (unsigned)(((r      ) * IN_F + c) * 2);
  unsigned offB1 = (unsigned)(((r +  64) * IN_F + c) * 2);

  // per-lane LDS dest byte addresses (buffer 0)
  const unsigned smemBase = (unsigned)(size_t)(void*)&smem[0];
  const unsigned dA  = smemBase + (unsigned)((r * LDA + c) * 2);
  const unsigned dB  = smemBase + (unsigned)(LDSA_HALVES * 2) + (unsigned)((r * LDA + c) * 2);
  const unsigned rowD = 64 * LDA * 2;     // 5120 B per 64-row group

  // fragment read pointers (buffer 0)
  const int fragRow = lane & 15;
  const int fragK   = (lane & 16) ? 8 : 0;
  const _Float16* fa0 = &smem[(wave * 32 + fragRow) * LDA + fragK];
  const _Float16* fb  = &smem[LDSA_HALVES + fragRow * LDA + fragK];

#define ISSUE_TILE(bufOff)                                            \
  do {                                                                \
    async_b128(dA + (bufOff),            offA0, sbaseA);              \
    async_b128(dA + (bufOff) +     rowD, offA1, sbaseA);              \
    async_b128(dA + (bufOff) + 2 * rowD, offA2, sbaseA);              \
    async_b128(dA + (bufOff) + 3 * rowD, offA3, sbaseA);              \
    async_b128(dB + (bufOff),            offB0, sbaseB);              \
    async_b128(dB + (bufOff) +     rowD, offB1, sbaseB);              \
    offA0 += 64; offA1 += 64; offA2 += 64; offA3 += 64;               \
    offB0 += 64; offB1 += 64;                                         \
  } while (0)

#define COMPUTE(bufOff)                                                               \
  do {                                                                                \
    const _Float16* pa = (const _Float16*)((const char*)fa0 + (bufOff));              \
    const _Float16* pb = (const _Float16*)((const char*)fb  + (bufOff));              \
    v16h a0 = LOADFRAG(pa);                                                           \
    v16h a1 = LOADFRAG(pa + 16 * LDA);                                                \
    v16h bcur = LOADFRAG(pb);                                                         \
    _Pragma("unroll")                                                                 \
    for (int f = 0; f < 8; ++f) {                                                     \
      v16h bnxt;                                                                      \
      if (f < 7) bnxt = LOADFRAG(pb + (f + 1) * 16 * LDA);                            \
      acc0[f] = __builtin_amdgcn_wmma_f32_16x16x32_f16(false, a0, false, bcur,        \
                                                       (short)0, acc0[f], false, false); \
      acc1[f] = __builtin_amdgcn_wmma_f32_16x16x32_f16(false, a1, false, bcur,        \
                                                       (short)0, acc1[f], false, false); \
      if (f < 7) bcur = bnxt;                                                         \
    }                                                                                 \
  } while (0)

  // prologue: tiles 0 and 1 in flight
  ISSUE_TILE(0u);
  ISSUE_TILE((unsigned)BUF_BYTES);

  unsigned b0 = 0, b1 = BUF_BYTES, b2 = 2 * BUF_BYTES;  // compute / next / fill
  for (int t = 0; t < NT - 1; ++t) {
    wait_async6();                 // tile t resident (t+1 may still be in flight)
    __syncthreads();               // all waves' tile t visible; b2 reads finished
    if (t + 2 < NT) ISSUE_TILE(b2);
    COMPUTE(b0);
    unsigned tmp = b0; b0 = b1; b1 = b2; b2 = tmp;
  }
  wait_async();                    // last tile fully resident
  __syncthreads();
  COMPUTE(b0);

#undef ISSUE_TILE
#undef COMPUTE

  // C/D layout: lanes 0-15: N=lane, M=v; lanes 16-31: N=lane-16, M=v+8
  const int n0 = tileN + (lane & 15);
  const int m0 = tileM + wave * 32 + ((lane & 16) ? 8 : 0);
#pragma unroll
  for (int f = 0; f < 8; ++f) {
    float* op0 = out + (size_t)m0 * OUT_F + n0 + f * 16;
    float* op1 = op0 + (size_t)16 * OUT_F;
#pragma unroll
    for (int v = 0; v < 8; ++v) {
      op0[(size_t)v * OUT_F] = acc0[f][v];
      op1[(size_t)v * OUT_F] = acc1[f][v];
    }
  }
}

extern "C" void kernel_launch(void* const* d_in, const int* in_sizes, int n_in,
                              void* d_out, int out_size, void* d_ws, size_t ws_size,
                              hipStream_t stream) {
  const float*    x    = (const float*)d_in[0];
  const _Float16* bw   = (const _Float16*)d_in[1];
  const _Float16* vals = (const _Float16*)d_in[2];
  const int*      rows = (const int*)d_in[3];
  const int*      cols = (const int*)d_in[4];
  float*          out  = (float*)d_out;

  const int nnz    = in_sizes[2];
  const int xElems = in_sizes[0];              // 8192 * 4096
  const int nRows  = xElems / IN_F;            // 8192

  // workspace layout: [0,32MB) W_comb fp16, [32MB,96MB) x fp16
  char* ws = (char*)d_ws;
  _Float16* wcomb = (_Float16*)ws;
  _Float16* x16   = (_Float16*)(ws + (size_t)OUT_F * IN_F * sizeof(_Float16));

  // 1) W_comb = base weights
  int wChunks = (OUT_F * IN_F) / 8;            // half8 chunks
  copy_w_kernel<<<wChunks / 256, 256, 0, stream>>>((const f4v*)bw, (f4v*)wcomb, wChunks);

  // 2) x -> fp16
  int x8 = xElems / 8;
  cvt_x_kernel<<<x8 / 256, 256, 0, stream>>>(x, x16, x8);

  // 3) scatter sparse ortho correction into W_comb
  scatter_kernel<<<(nnz + 255) / 256, 256, 0, stream>>>(vals, rows, cols,
                                                        (unsigned int*)wcomb, nnz);

  // 4) out = x16 @ W_comb^T  (f16 WMMA, async-LDS triple-buffered pipeline)
  dim3 grid(OUT_F / BN, nRows / BM);
  gemm_wmma_kernel<<<grid, 256, 0, stream>>>(x16, wcomb, out);
}
